// transformer_63247688401494
// MI455X (gfx1250) — matrix operand order repaired and checked by reading, not compiled, not run on hardware
//
#include <hip/hip_runtime.h>

#define DIMS  1024
#define DEPTH 8
#define HEADS 16
#define DHEAD 64
#define BSZ   2
#define SEQ   2048
#define MTOK  (BSZ*SEQ)        // 4096 tokens
#define FFD   (DIMS*4)         // 4096
#define CPD   (DHEAD*DEPTH)    // 512
#define QKVN  (3*HEADS*DHEAD)  // 3072

typedef __attribute__((ext_vector_type(16))) __bf16 v16bf;
typedef __attribute__((ext_vector_type(8)))  float  v8f;

// ---- CDNA5 async global->LDS path (ASYNCcnt), with compile-safe fallback ----
#if defined(__has_builtin)
#if __has_builtin(__builtin_amdgcn_global_load_async_to_lds_b128)
#define HAVE_ASYNC_LDS 1
#endif
#endif

#ifdef HAVE_ASYNC_LDS
typedef int int4g __attribute__((vector_size(16)));
typedef __attribute__((address_space(1))) int4g as1_int4;
typedef __attribute__((address_space(3))) int4g as3_int4;
#define ASYNC_B128(ldsp, gp)                                                   \
  __builtin_amdgcn_global_load_async_to_lds_b128((as1_int4*)(gp),              \
                                                 (as3_int4*)(ldsp), 0, 0)
#if __has_builtin(__builtin_amdgcn_s_wait_asynccnt)
#define WAIT_ASYNC() __builtin_amdgcn_s_wait_asynccnt(0)
#else
#define WAIT_ASYNC() asm volatile("s_wait_asynccnt 0" ::: "memory")
#endif
#else
#define WAIT_ASYNC()
#endif

union Frag {
  v16bf v;
  uint4 u4[2];
  unsigned short us[16];
};

__device__ __forceinline__ unsigned short f2bf(float f) {
  unsigned int u = __float_as_uint(f);
  u += 0x7FFFu + ((u >> 16) & 1u);          // round-to-nearest-even
  return (unsigned short)(u >> 16);
}
__device__ __forceinline__ float bf2f(unsigned short h) {
  return __uint_as_float(((unsigned int)h) << 16);
}
__device__ __forceinline__ v8f v8zero() {
  v8f z = {0.f,0.f,0.f,0.f,0.f,0.f,0.f,0.f};
  return z;
}

// ---------------- elementwise helpers ----------------
__global__ void k_f32_to_bf16(unsigned short* __restrict__ o,
                              const float* __restrict__ i, int n) {
  int idx = blockIdx.x * 256 + threadIdx.x;
  if (idx < n) o[idx] = f2bf(i[idx]);
}

__global__ void k_layernorm(unsigned short* __restrict__ out,
                            const float* __restrict__ in,
                            const float* __restrict__ g,
                            const float* __restrict__ b, int cols) {
  __shared__ float s1[256], s2[256];
  int row = blockIdx.x, tid = threadIdx.x;
  const float* r = in + (size_t)row * cols;
  float a = 0.f, q = 0.f;
  for (int c = tid; c < cols; c += 256) { float v = r[c]; a += v; q += v * v; }
  s1[tid] = a; s2[tid] = q; __syncthreads();
  for (int st = 128; st; st >>= 1) {
    if (tid < st) { s1[tid] += s1[tid + st]; s2[tid] += s2[tid + st]; }
    __syncthreads();
  }
  float mean = s1[0] / cols;
  float var  = s2[0] / cols - mean * mean;
  float inv  = rsqrtf(var + 1e-5f);
  unsigned short* w = out + (size_t)row * cols;
  for (int c = tid; c < cols; c += 256)
    w[c] = f2bf((r[c] - mean) * inv * g[c] + b[c]);
}

// qkv flat [MTOK, 3072] bf16, inner ordering (h, d, s) -> split to [B,H,N,D] bf16
__global__ void k_split_qkv(unsigned short* __restrict__ q,
                            unsigned short* __restrict__ kk,
                            unsigned short* __restrict__ vv,
                            float* __restrict__ kv_last,
                            const unsigned short* __restrict__ qkv, int layer) {
  int idx = blockIdx.x * 256 + threadIdx.x;
  if (idx >= MTOK * QKVN) return;
  int m = idx / QKVN;
  int n = idx % QKVN;
  int h = n / (DHEAD * 3);
  int r = n % (DHEAD * 3);
  int d = r / 3, s = r % 3;
  int b = m / SEQ, tok = m % SEQ;
  unsigned short val = qkv[idx];
  unsigned short* dst = (s == 0) ? q : (s == 1) ? kk : vv;
  dst[(((size_t)(b * HEADS + h)) * SEQ + tok) * DHEAD + d] = val;
  if (tok == SEQ - 1 && s > 0)
    kv_last[(((size_t)(s - 1) * BSZ + b) * HEADS + h) * CPD + layer * DHEAD + d] = bf2f(val);
}

__global__ void k_glu(unsigned short* __restrict__ out,
                      const unsigned short* __restrict__ u,
                      int half, int siluFirst, int n) {
  int idx = blockIdx.x * 256 + threadIdx.x;
  if (idx >= n) return;
  int m = idx / half, j = idx % half;
  float a = bf2f(u[(size_t)m * 2 * half + j]);
  float g = bf2f(u[(size_t)m * 2 * half + half + j]);
  float o = siluFirst ? (a / (1.f + __expf(-a))) * g
                      : a * (g / (1.f + __expf(-g)));
  out[idx] = f2bf(o);
}

// cache [2,B,H,1,CPD] f32 -> out [L,2,B,H,1,DHEAD]
__global__ void k_cache_out(float* __restrict__ out, const float* __restrict__ c) {
  int idx = blockIdx.x * 256 + threadIdx.x;
  if (idx >= DEPTH * 2 * BSZ * HEADS * DHEAD) return;
  int d = idx % DHEAD; int t = idx / DHEAD;
  int h = t % HEADS;   t /= HEADS;
  int b = t % BSZ;     t /= BSZ;
  int s = t % 2;       int l = t / 2;
  out[idx] = c[(((size_t)s * BSZ + b) * HEADS + h) * CPD + l * DHEAD + d];
}

// ---------------- WMMA GEMM: C[M,N] = A[M,K] * B[N,K]^T (+bias)(+res) ----------------
// 256 threads = 8 waves; block tile 128x128, BK=64; wave tile 32x64 = 2x4 WMMA tiles.
// Double-buffered LDS; tile t+1 streamed via GLOBAL_LOAD_ASYNC_TO_LDS while tile t
// feeds the WMMAs (falls back to synchronous staging if the builtin is absent).
__global__ __launch_bounds__(256) void k_gemm_bf16(
    const unsigned short* __restrict__ A,   // [M,K] bf16 row-major
    const unsigned short* __restrict__ Bw,  // [N,K] bf16 row-major (acts transposed)
    float* __restrict__ Cf, unsigned short* __restrict__ Cb,
    const float* __restrict__ bias, const float* __restrict__ res,
    int M, int N, int K, int storeBf) {
  __shared__ __align__(16) unsigned short As[2][128 * 72];  // pad: 64+8 halfs
  __shared__ __align__(16) unsigned short Bs[2][128 * 72];
  int tid = threadIdx.x, lane = tid & 31, wid = tid >> 5;
  int wm = (wid & 3) * 32, wn = (wid >> 2) * 64;
  int bm = blockIdx.y * 128, bn = blockIdx.x * 128;
  int lhi = (lane & 16) ? 1 : 0;
  v8f acc[2][4];
#pragma unroll
  for (int mt = 0; mt < 2; ++mt)
#pragma unroll
    for (int nt = 0; nt < 4; ++nt) acc[mt][nt] = v8zero();

  int lrow = tid >> 1, lcol = (tid & 1) * 32;   // each thread stages 32 halfs
  bool aok = (bm + lrow) < M;
  if (!aok) {                                    // padded rows: zero once, never rewritten
    uint4 z = {0, 0, 0, 0};
    uint4* s0 = (uint4*)(As[0] + lrow * 72 + lcol);
    uint4* s1 = (uint4*)(As[1] + lrow * 72 + lcol);
    s0[0] = z; s0[1] = z; s0[2] = z; s0[3] = z;
    s1[0] = z; s1[1] = z; s1[2] = z; s1[3] = z;
  }

  auto stage = [&](int k0, int bi) {
    if (aok) {
      const unsigned short* ga = A + (size_t)(bm + lrow) * K + k0 + lcol;
      unsigned short* sa = As[bi] + lrow * 72 + lcol;
#ifdef HAVE_ASYNC_LDS
      ASYNC_B128(sa, ga);           ASYNC_B128(sa + 8, ga + 8);
      ASYNC_B128(sa + 16, ga + 16); ASYNC_B128(sa + 24, ga + 24);
#else
      ((uint4*)sa)[0] = ((const uint4*)ga)[0];
      ((uint4*)sa)[1] = ((const uint4*)ga)[1];
      ((uint4*)sa)[2] = ((const uint4*)ga)[2];
      ((uint4*)sa)[3] = ((const uint4*)ga)[3];
#endif
    }
    const unsigned short* gb = Bw + (size_t)(bn + lrow) * K + k0 + lcol;
    unsigned short* sb = Bs[bi] + lrow * 72 + lcol;
#ifdef HAVE_ASYNC_LDS
    ASYNC_B128(sb, gb);           ASYNC_B128(sb + 8, gb + 8);
    ASYNC_B128(sb + 16, gb + 16); ASYNC_B128(sb + 24, gb + 24);
#else
    ((uint4*)sb)[0] = ((const uint4*)gb)[0];
    ((uint4*)sb)[1] = ((const uint4*)gb)[1];
    ((uint4*)sb)[2] = ((const uint4*)gb)[2];
    ((uint4*)sb)[3] = ((const uint4*)gb)[3];
#endif
  };

  stage(0, 0);
  WAIT_ASYNC();
  __syncthreads();

  int T = K / 64;
  for (int t = 0; t < T; ++t) {
    int bi = t & 1;
    if (t + 1 < T) stage((t + 1) * 64, bi ^ 1);   // overlap with compute below
    const unsigned short* Ab = As[bi];
    const unsigned short* Bb = Bs[bi];
#pragma unroll
    for (int c = 0; c < 2; ++c) {
      Frag af[2], bfr[4];
      int klo = c * 32 + (lhi ? 8 : 0);           // A layout: lanes<16 -> K 0..7,16..23
#pragma unroll
      for (int mt = 0; mt < 2; ++mt) {
        const unsigned short* p = Ab + (wm + mt * 16 + (lane & 15)) * 72 + klo;
        af[mt].u4[0] = *(const uint4*)p;
        af[mt].u4[1] = *(const uint4*)(p + 16);
      }
      int kb = c * 32 + (lhi ? 8 : 0);            // B layout: lanes<16 -> K 0..7,16..23
#pragma unroll
      for (int nt = 0; nt < 4; ++nt) {
        const uint4* p = (const uint4*)(Bb + (wn + nt * 16 + (lane & 15)) * 72 + kb);
        bfr[nt].u4[0] = p[0]; bfr[nt].u4[1] = p[2];
      }
#pragma unroll
      for (int mt = 0; mt < 2; ++mt)
#pragma unroll
        for (int nt = 0; nt < 4; ++nt)
          acc[mt][nt] = __builtin_amdgcn_wmma_f32_16x16x32_bf16(
              false, af[mt].v, false, bfr[nt].v, (short)0, acc[mt][nt], false, false);
    }
    WAIT_ASYNC();
    __syncthreads();
  }

#pragma unroll
  for (int mt = 0; mt < 2; ++mt)
#pragma unroll
    for (int nt = 0; nt < 4; ++nt)
#pragma unroll
      for (int e = 0; e < 8; ++e) {
        int row = bm + wm + mt * 16 + (lhi ? e + 8 : e);
        int col = bn + wn + nt * 16 + (lane & 15);
        if (row < M) {
          float vv = acc[mt][nt][e];
          if (bias) vv += bias[col];
          if (res)  vv += res[(size_t)row * N + col];
          if (storeBf) Cb[(size_t)row * N + col] = f2bf(vv);
          else         Cf[(size_t)row * N + col] = vv;
        }
      }
}

// ---------------- flash attention: 4 waves, 16 q-rows/wave, 32-key tiles ----------------
__global__ __launch_bounds__(128) void k_attention(
    unsigned short* __restrict__ o,          // [MTOK, H*D] bf16
    const unsigned short* __restrict__ q,    // [B*H, N, D] bf16
    const unsigned short* __restrict__ k,
    const unsigned short* __restrict__ v) {
  __shared__ __align__(16) unsigned short Ks[32 * 72];     // [key][d], pad 8
  __shared__ __align__(16) unsigned short Vt[64 * 40];     // [d][key], pad 8
  __shared__ __align__(16) unsigned short Pw[4 * 16 * 40]; // wave-private P
  int tid = threadIdx.x, lane = tid & 31, wid = tid >> 5;
  int bh = blockIdx.y;
  int qbase = blockIdx.x * 64 + wid * 16;
  int b = bh >> 4, h = bh & 15;
  int lhi = (lane & 16) ? 1 : 0;

  Frag qf[2];
  const unsigned short* qrow = q + (((size_t)bh * SEQ) + qbase + (lane & 15)) * DHEAD;
#pragma unroll
  for (int c = 0; c < 2; ++c) {
    int klo = c * 32 + (lhi ? 8 : 0);
    qf[c].u4[0] = *(const uint4*)(qrow + klo);
    qf[c].u4[1] = *(const uint4*)(qrow + klo + 16);
  }

  float mrow[8], lrow[8];
#pragma unroll
  for (int e = 0; e < 8; ++e) { mrow[e] = -1e30f; lrow[e] = 0.f; }
  v8f accO[4];
#pragma unroll
  for (int dt = 0; dt < 4; ++dt) accO[dt] = v8zero();

  int jrow = tid >> 2, jcol = (tid & 3) * 16;
  for (int j0 = 0; j0 < SEQ; j0 += 32) {
    // stage K tile row-major
    const uint4* kg = (const uint4*)(k + (((size_t)bh * SEQ) + j0 + jrow) * DHEAD + jcol);
    uint4* ks = (uint4*)(Ks + jrow * 72 + jcol);
    ks[0] = kg[0]; ks[1] = kg[1];
    // stage V transposed [d][key]
    Frag tv;
    const uint4* vg = (const uint4*)(v + (((size_t)bh * SEQ) + j0 + jrow) * DHEAD + jcol);
    tv.u4[0] = vg[0]; tv.u4[1] = vg[1];
#pragma unroll
    for (int i = 0; i < 16; ++i) Vt[(jcol + i) * 40 + jrow] = tv.us[i];
    __syncthreads();

    // S = Q K^T  (two 16-key sub-tiles)
    v8f s0 = v8zero(), s1 = v8zero();
#pragma unroll
    for (int c = 0; c < 2; ++c) {
      Frag kf;
      int kb = c * 32 + (lhi ? 8 : 0);
      const uint4* p0 = (const uint4*)(Ks + (lane & 15) * 72 + kb);
      kf.u4[0] = p0[0]; kf.u4[1] = p0[2];
      s0 = __builtin_amdgcn_wmma_f32_16x16x32_bf16(false, qf[c].v, false, kf.v, (short)0, s0, false, false);
      const uint4* p1 = (const uint4*)(Ks + (16 + (lane & 15)) * 72 + kb);
      kf.u4[0] = p1[0]; kf.u4[1] = p1[2];
      s1 = __builtin_amdgcn_wmma_f32_16x16x32_bf16(false, qf[c].v, false, kf.v, (short)0, s1, false, false);
    }

    // online softmax (row spread over 16 lanes; reduce with xor<16)
    float pe0[8], pe1[8], corr[8];
#pragma unroll
    for (int e = 0; e < 8; ++e) {
      float a  = s0[e] * 0.125f;   // DHEAD^-0.5
      float bb = s1[e] * 0.125f;
      float mx = fmaxf(a, bb);
#pragma unroll
      for (int off = 1; off < 16; off <<= 1) mx = fmaxf(mx, __shfl_xor(mx, off, 32));
      float mn = fmaxf(mrow[e], mx);
      float c0 = __expf(mrow[e] - mn);
      float p0 = __expf(a - mn), p1 = __expf(bb - mn);
      float sum = p0 + p1;
#pragma unroll
      for (int off = 1; off < 16; off <<= 1) sum += __shfl_xor(sum, off, 32);
      lrow[e] = lrow[e] * c0 + sum;
      mrow[e] = mn; corr[e] = c0; pe0[e] = p0; pe1[e] = p1;
    }
#pragma unroll
    for (int dt = 0; dt < 4; ++dt)
#pragma unroll
      for (int e = 0; e < 8; ++e) accO[dt][e] *= corr[e];

    // P (C-layout) -> wave-private LDS -> A-layout fragment
    unsigned short* pw = Pw + wid * 16 * 40;
#pragma unroll
    for (int e = 0; e < 8; ++e) {
      int r = lhi ? e + 8 : e;
      pw[r * 40 + (lane & 15)]      = f2bf(pe0[e]);
      pw[r * 40 + 16 + (lane & 15)] = f2bf(pe1[e]);
    }
    asm volatile("s_wait_dscnt 0" ::: "memory");   // same-wave LDS WAR/RAW fence
    Frag pf;
    {
      const unsigned short* pr = pw + (lane & 15) * 40 + (lhi ? 8 : 0);
      pf.u4[0] = *(const uint4*)(pr);
      pf.u4[1] = *(const uint4*)(pr + 16);
    }
    // O += P V
#pragma unroll
    for (int dt = 0; dt < 4; ++dt) {
      Frag vf;
      int d = dt * 16 + (lane & 15);
      const uint4* p = (const uint4*)(Vt + d * 40 + (lhi ? 8 : 0));
      vf.u4[0] = p[0]; vf.u4[1] = p[2];
      accO[dt] = __builtin_amdgcn_wmma_f32_16x16x32_bf16(false, pf.v, false, vf.v, (short)0, accO[dt], false, false);
    }
    __syncthreads();
  }

#pragma unroll
  for (int dt = 0; dt < 4; ++dt)
#pragma unroll
    for (int e = 0; e < 8; ++e) {
      int r = lhi ? e + 8 : e;
      int tok = qbase + r;
      int col = h * DHEAD + dt * 16 + (lane & 15);
      o[((size_t)(b * SEQ + tok)) * (HEADS * DHEAD) + col] = f2bf(accO[dt][e] / lrow[e]);
    }
}

// ---------------- host orchestration ----------------
extern "C" void kernel_launch(void* const* d_in, const int* in_sizes, int n_in,
                              void* d_out, int out_size, void* d_ws, size_t ws_size,
                              hipStream_t stream) {
  (void)in_sizes; (void)n_in; (void)out_size; (void)ws_size;
  const float* x_in = (const float*)d_in[0];
  char* ws = (char*)d_ws;
  const size_t MB = 1ull << 20;

  float*          x_ws   = (float*)(ws + 0);                    // 16 MB
  unsigned short* h_bf   = (unsigned short*)(ws + 16 * MB);     // 8 MB
  char*           big    = ws + 24 * MB;                        // 64 MB union region
  unsigned short* qkv_bf = (unsigned short*)(big);              // 24 MB
  unsigned short* q_bf   = (unsigned short*)(big + 24 * MB);    // 8 MB
  unsigned short* k_bf   = (unsigned short*)(big + 32 * MB);    // 8 MB
  unsigned short* v_bf   = (unsigned short*)(big + 40 * MB);    // 8 MB
  unsigned short* o_bf   = (unsigned short*)(big);              // aliases dead qkv_bf
  unsigned short* u_bf   = (unsigned short*)(big);              // aliases dead q/k/v/o
  unsigned short* glu_bf = (unsigned short*)(ws + 88 * MB);     // 32 MB
  unsigned short* qkvw   = (unsigned short*)(ws + 120 * MB);    // 6 MB
  unsigned short* outw   = (unsigned short*)(ws + 126 * MB);    // 2 MB
  unsigned short* gluw   = (unsigned short*)(ws + 128 * MB);    // 32 MB
  unsigned short* ff2w   = (unsigned short*)(ws + 160 * MB);    // 8 MB
  unsigned short* cpw1   = (unsigned short*)(ws + 168 * MB);    // 2 MB
  unsigned short* cpw2   = (unsigned short*)(ws + 170 * MB);    // 1 MB
  float*          kvlast = (float*)(ws + 171 * MB);
  unsigned short* kvln   = (unsigned short*)(ws + 171 * MB + 256 * 1024);
  unsigned short* cpu_bf = (unsigned short*)(ws + 171 * MB + 512 * 1024);
  unsigned short* cpact  = (unsigned short*)(ws + 172 * MB);
  float*          cpc    = (float*)(ws + 172 * MB + 256 * 1024);

  (void)hipMemcpyAsync(x_ws, x_in, (size_t)MTOK * DIMS * sizeof(float),
                       hipMemcpyDeviceToDevice, stream);

  for (int l = 0; l < DEPTH; ++l) {
    const float* ln1g = (const float*)d_in[1 + l * 10 + 0];
    const float* ln1b = (const float*)d_in[1 + l * 10 + 1];
    const float* qkvW = (const float*)d_in[1 + l * 10 + 2];
    const float* outW = (const float*)d_in[1 + l * 10 + 3];
    const float* ln2g = (const float*)d_in[1 + l * 10 + 4];
    const float* ln2b = (const float*)d_in[1 + l * 10 + 5];
    const float* gluW = (const float*)d_in[1 + l * 10 + 6];
    const float* gluB = (const float*)d_in[1 + l * 10 + 7];
    const float* ff2W = (const float*)d_in[1 + l * 10 + 8];
    const float* ff2B = (const float*)d_in[1 + l * 10 + 9];

    k_f32_to_bf16<<<(QKVN * DIMS + 255) / 256, 256, 0, stream>>>(qkvw, qkvW, QKVN * DIMS);
    k_f32_to_bf16<<<(DIMS * DIMS + 255) / 256, 256, 0, stream>>>(outw, outW, DIMS * DIMS);
    k_f32_to_bf16<<<(2 * FFD * DIMS + 255) / 256, 256, 0, stream>>>(gluw, gluW, 2 * FFD * DIMS);
    k_f32_to_bf16<<<(DIMS * FFD + 255) / 256, 256, 0, stream>>>(ff2w, ff2W, DIMS * FFD);

    k_layernorm<<<MTOK, 256, 0, stream>>>(h_bf, x_ws, ln1g, ln1b, DIMS);
    {
      dim3 g(QKVN / 128, MTOK / 128);
      k_gemm_bf16<<<g, 256, 0, stream>>>(h_bf, qkvw, nullptr, qkv_bf, nullptr, nullptr,
                                         MTOK, QKVN, DIMS, 1);
    }
    k_split_qkv<<<(MTOK * QKVN + 255) / 256, 256, 0, stream>>>(q_bf, k_bf, v_bf, kvlast, qkv_bf, l);
    {
      dim3 g(SEQ / 64, BSZ * HEADS);
      k_attention<<<g, 128, 0, stream>>>(o_bf, q_bf, k_bf, v_bf);
    }
    {
      dim3 g(DIMS / 128, MTOK / 128);
      k_gemm_bf16<<<g, 256, 0, stream>>>(o_bf, outw, x_ws, nullptr, nullptr, x_ws,
                                         MTOK, DIMS, HEADS * DHEAD, 0);
    }
    k_layernorm<<<MTOK, 256, 0, stream>>>(h_bf, x_ws, ln2g, ln2b, DIMS);
    {
      dim3 g(2 * FFD / 128, MTOK / 128);
      k_gemm_bf16<<<g, 256, 0, stream>>>(h_bf, gluw, nullptr, u_bf, gluB, nullptr,
                                         MTOK, 2 * FFD, DIMS, 1);
    }
    k_glu<<<(MTOK * FFD + 255) / 256, 256, 0, stream>>>(glu_bf, u_bf, FFD, 0, MTOK * FFD);
    {
      dim3 g(DIMS / 128, MTOK / 128);
      k_gemm_bf16<<<g, 256, 0, stream>>>(glu_bf, ff2w, x_ws, nullptr, ff2B, x_ws,
                                         MTOK, DIMS, FFD, 0);
    }
  }

  // cache projection over last-token kv  (64 rows of 512)
  const float* cplng = (const float*)d_in[1 + DEPTH * 10 + 0];
  const float* cplnb = (const float*)d_in[1 + DEPTH * 10 + 1];
  const float* cpW1  = (const float*)d_in[1 + DEPTH * 10 + 2];
  const float* cpB1  = (const float*)d_in[1 + DEPTH * 10 + 3];
  const float* cpW2  = (const float*)d_in[1 + DEPTH * 10 + 4];
  const float* cpB2  = (const float*)d_in[1 + DEPTH * 10 + 5];
  k_f32_to_bf16<<<(CPD * 4 * CPD + 255) / 256, 256, 0, stream>>>(cpw1, cpW1, CPD * 4 * CPD);
  k_f32_to_bf16<<<(CPD * 2 * CPD + 255) / 256, 256, 0, stream>>>(cpw2, cpW2, CPD * 2 * CPD);
  k_layernorm<<<2 * BSZ * HEADS, 256, 0, stream>>>(kvln, kvlast, cplng, cplnb, CPD);
  {
    dim3 g(CPD * 4 / 128, 1);
    k_gemm_bf16<<<g, 256, 0, stream>>>(kvln, cpw1, nullptr, cpu_bf, cpB1, nullptr,
                                       2 * BSZ * HEADS, CPD * 4, CPD, 1);
  }
  k_glu<<<(2 * BSZ * HEADS * 2 * CPD + 255) / 256, 256, 0, stream>>>(
      cpact, cpu_bf, 2 * CPD, 1, 2 * BSZ * HEADS * 2 * CPD);
  {
    dim3 g(CPD / 128, 1);
    k_gemm_bf16<<<g, 256, 0, stream>>>(cpact, cpw2, cpc, nullptr, cpB2, nullptr,
                                       2 * BSZ * HEADS, CPD, 2 * CPD, 0);
  }

  float* out = (float*)d_out;
  (void)hipMemcpyAsync(out, x_ws, (size_t)MTOK * DIMS * sizeof(float),
                       hipMemcpyDeviceToDevice, stream);
  k_cache_out<<<(DEPTH * 2 * BSZ * HEADS * DHEAD + 255) / 256, 256, 0, stream>>>(
      out + (size_t)MTOK * DIMS, cpc);
}
